// BasicGCNRegressor_66425964200347
// MI455X (gfx1250) — matrix operand order repaired and checked by reading, not compiled, not run on hardware
//
#include <hip/hip_runtime.h>
#include <hip/hip_bf16.h>

#define HID 128
#define NG 8

typedef __attribute__((ext_vector_type(16))) __bf16 v16bf;
typedef __attribute__((ext_vector_type(8)))  float  v8f;

union BfVec { v16bf v; unsigned short u[16]; };

__device__ __forceinline__ unsigned short f2bf(float f) {
    // round-to-nearest-even f32 -> bf16 bit pattern
    unsigned int u = __float_as_uint(f);
    unsigned int r = u + 0x7FFFu + ((u >> 16) & 1u);
    return (unsigned short)(r >> 16);
}

// ---------------- degree kernels ----------------
__global__ void deg_kernel(const int* __restrict__ src, const int* __restrict__ dst,
                           float* __restrict__ dout, float* __restrict__ din, int nE) {
    int e = blockIdx.x * blockDim.x + threadIdx.x;
    if (e < nE) {
        atomicAdd(&dout[src[e]], 1.0f);
        atomicAdd(&din[dst[e]], 1.0f);
    }
}

__global__ void isqrt_kernel(float* __restrict__ d, int n) {
    int i = blockIdx.x * blockDim.x + threadIdx.x;
    if (i < n) d[i] = rsqrtf(fmaxf(d[i], 1.0f));
}

// ---------------- SpMM (gather + scatter-add), fused out-degree scale ----------------
__global__ void spmm5_kernel(const int* __restrict__ src, const int* __restrict__ dst,
                             const float* __restrict__ feat, const float* __restrict__ osq,
                             float* __restrict__ agg5, int nE) {
    int e = blockIdx.x * blockDim.x + threadIdx.x;
    if (e < nE) {
        int s = src[e], d = dst[e];
        float sc = osq[s];
        #pragma unroll
        for (int c = 0; c < 5; ++c)
            atomicAdd(&agg5[(size_t)d * 5 + c], feat[(size_t)s * 5 + c] * sc);
    }
}

__global__ void spmm128_kernel(const int* __restrict__ src, const int* __restrict__ dst,
                               const float* __restrict__ h, const float* __restrict__ osq,
                               float* __restrict__ agg, int nE) {
    long long idx = (long long)blockIdx.x * blockDim.x + threadIdx.x;
    int e  = (int)(idx >> 5);          // 32 threads per edge
    int c4 = (int)(idx & 31) * 4;      // 4 floats per thread
    if (e >= nE) return;
    int s = src[e], d = dst[e];
    float sc = osq[s];
    const float4 hv = *(const float4*)(h + (size_t)s * HID + c4);
    float* ap = agg + (size_t)d * HID + c4;
    atomicAdd(ap + 0, hv.x * sc);
    atomicAdd(ap + 1, hv.y * sc);
    atomicAdd(ap + 2, hv.z * sc);
    atomicAdd(ap + 3, hv.w * sc);
}

// ---------------- layer 1 GEMM (K=5, VALU) ----------------
__global__ void layer1_kernel(const float* __restrict__ agg5, const float* __restrict__ isq,
                              const float* __restrict__ W1, const float* __restrict__ b1,
                              float* __restrict__ h, int nN) {
    long long idx = (long long)blockIdx.x * blockDim.x + threadIdx.x;
    int n = (int)(idx >> 7);
    int j = (int)(idx & 127);
    if (n >= nN) return;
    float s = 0.0f;
    #pragma unroll
    for (int k = 0; k < 5; ++k)
        s += agg5[(size_t)n * 5 + k] * W1[k * HID + j];
    s = s * isq[n] + b1[j];
    h[(size_t)n * HID + j] = s > 0.0f ? s : 0.0f;
}

// ---------------- WMMA GEMM + bias + ReLU, fused in-degree scale ----------------
// block = 256 threads = 8 waves. Block covers 64 rows x 128 cols.
// W staged TRANSPOSED in LDS (Wt[n][k]): each lane's 16 B values contiguous
// -> 2x ds_load_b128; B fragments register-cached across the 4 row tiles.
// Inner loop: 2x ds_load_b128 (A) + v_wmma_f32_16x16x32_bf16.
// Epilogue: scalar-uniform full-tile test -> branch-free stores with constant
// immediate offsets for all but the final (partial) block.
__global__ __launch_bounds__(256) void gemm_relu_wmma(
        const float* __restrict__ agg, const float* __restrict__ isq,
        const float* __restrict__ W, const float* __restrict__ bias,
        float* __restrict__ out, int nN) {
    __shared__ unsigned short Wt[HID * 136];   // bf16 bits, [n][k] transposed, padded
    __shared__ unsigned short As[64 * 136];    // bf16 bits, [r][k], padded

    const int tid = threadIdx.x;
    const int row0 = blockIdx.x * 64;

    // stage W (f32 -> bf16), transposed, into LDS
    for (int i = tid; i < HID * HID; i += 256) {
        int k = i >> 7, n = i & 127;           // W row-major [k][n]
        Wt[n * 136 + k] = f2bf(W[i]);
    }
    // stage 64-row A tile, scaled by in_isqrt, into LDS
    for (int i = tid; i < 64 * HID; i += 256) {
        int r = i >> 7, k = i & 127;
        int gr = row0 + r;
        float x = 0.0f;
        if (gr < nN) x = agg[(size_t)gr * HID + k] * isq[gr];
        As[r * 136 + k] = f2bf(x);
    }
    __syncthreads();

    const int wave = tid >> 5;
    const int lane = tid & 31;
    const int n0   = wave * 16;      // column tile
    const int m    = lane & 15;
    const int half = lane >> 4;
    const float bval = bias[n0 + m];
    const bool full = (row0 + 64 <= nN);   // uniform across block

    // register-cache all 4 B fragments (K=32 each) for this wave's column tile.
    // B 32x16 layout: lanes 0-15 hold K=0..15 (b.u[j] = W[k0+j][n]), lanes 16-31 K=16..31.
    BfVec bfr[4];
    #pragma unroll
    for (int kt = 0; kt < 4; ++kt) {
        const unsigned short* bp = &Wt[(n0 + m) * 136 + kt * 32 + 16 * half];
        #pragma unroll
        for (int j = 0; j < 16; ++j) bfr[kt].u[j] = bp[j];
    }

    for (int rt = 0; rt < 4; ++rt) {
        v8f acc = {};
        #pragma unroll
        for (int kt = 0; kt < 4; ++kt) {
            const int k0 = kt * 32;
            // A 16x32 bf16 layout: a.u[0..7]  = As_row[k0 + 16*half + 0..7]
            //                      a.u[8..15] = As_row[k0 + 16*half + 8..15]
            BfVec a;
            const unsigned short* ap = &As[(rt * 16 + m) * 136 + k0 + 16 * half];
            #pragma unroll
            for (int j = 0; j < 8; ++j) a.u[j] = ap[j];
            #pragma unroll
            for (int j = 0; j < 8; ++j) a.u[8 + j] = ap[8 + j];
            acc = __builtin_amdgcn_wmma_f32_16x16x32_bf16(
                      false, a.v, false, bfr[kt].v, (short)0, acc, false, false);
        }
        // D layout: acc[i] -> row (rt*16 + i + 8*half), col (n0 + m)
        float* op = out + (size_t)(row0 + rt * 16 + 8 * half) * HID + n0 + m;
        if (full) {
            #pragma unroll
            for (int i = 0; i < 8; ++i) {
                float x = acc[i] + bval;
                op[i * HID] = x > 0.0f ? x : 0.0f;   // constant-offset stores
            }
        } else {
            #pragma unroll
            for (int i = 0; i < 8; ++i) {
                int gr = row0 + rt * 16 + i + 8 * half;
                if (gr < nN) {
                    float x = acc[i] + bval;
                    op[i * HID] = x > 0.0f ? x : 0.0f;
                }
            }
        }
    }
}

// ---------------- pooling (graph_ids sorted -> LDS partials) ----------------
__global__ __launch_bounds__(128) void pool_kernel(const float* __restrict__ h,
                                                   const int* __restrict__ gid,
                                                   float* __restrict__ gsums,
                                                   float* __restrict__ cnt, int nN) {
    __shared__ float ls[NG][HID];
    __shared__ float lc[NG];
    int tid = threadIdx.x;
    #pragma unroll
    for (int g = 0; g < NG; ++g) ls[g][tid] = 0.0f;
    if (tid < NG) lc[tid] = 0.0f;
    __syncthreads();
    int base = blockIdx.x * 128;
    for (int i = 0; i < 128; ++i) {
        int n = base + i;
        if (n >= nN) break;                 // uniform across block
        int g = gid[n];
        ls[g][tid] += h[(size_t)n * HID + tid];
        if (tid == 0) lc[g] += 1.0f;
    }
    __syncthreads();
    #pragma unroll
    for (int g = 0; g < NG; ++g) atomicAdd(&gsums[g * HID + tid], ls[g][tid]);
    if (tid < NG) atomicAdd(&cnt[tid], lc[tid]);
}

// ---------------- final projection: 8 graphs x dot(128) ----------------
__global__ void final_kernel(const float* __restrict__ gsums, const float* __restrict__ cnt,
                             const float* __restrict__ Wp, const float* __restrict__ bp,
                             float* __restrict__ out) {
    int g = threadIdx.x >> 5;   // 256 threads -> 8 waves, one per graph
    int lane = threadIdx.x & 31;
    float c = fmaxf(cnt[g], 1.0f);
    float p = 0.0f;
    for (int j = lane; j < HID; j += 32)
        p += (gsums[g * HID + j] / c) * Wp[j];
    #pragma unroll
    for (int o = 16; o > 0; o >>= 1) p += __shfl_down(p, o, 32);
    if (lane == 0) out[g] = p + bp[0];
}

extern "C" void kernel_launch(void* const* d_in, const int* in_sizes, int n_in,
                              void* d_out, int out_size, void* d_ws, size_t ws_size,
                              hipStream_t stream) {
    const float* features = (const float*)d_in[0];
    const int*   src      = (const int*)d_in[1];
    const int*   dst      = (const int*)d_in[2];
    const int*   gids     = (const int*)d_in[3];
    const float* W1 = (const float*)d_in[4];  const float* b1 = (const float*)d_in[5];
    const float* W2 = (const float*)d_in[6];  const float* b2 = (const float*)d_in[7];
    const float* W3 = (const float*)d_in[8];  const float* b3 = (const float*)d_in[9];
    const float* W4 = (const float*)d_in[10]; const float* b4 = (const float*)d_in[11];
    const float* Wp = (const float*)d_in[12]; const float* bp = (const float*)d_in[13];
    float* out = (float*)d_out;

    const int N = in_sizes[3];      // graph_ids length = N_NODES
    const int E = in_sizes[1];      // src length = N_EDGES

    // workspace layout (floats)
    float* ws = (float*)d_ws;
    float* deg_out = ws;                               // N
    float* deg_in  = deg_out + N;                      // N
    float* agg5    = deg_in + N;                       // 5N
    float* h       = agg5 + (size_t)5 * N;             // 128N
    float* agg     = h + (size_t)HID * N;              // 128N
    float* gsums   = agg + (size_t)HID * N;            // NG*HID
    float* cnt     = gsums + NG * HID;                 // NG

    const int T = 256;

    // degrees -> rsqrt
    hipMemsetAsync(deg_out, 0, (size_t)2 * N * sizeof(float), stream);
    deg_kernel<<<(E + T - 1) / T, T, 0, stream>>>(src, dst, deg_out, deg_in, E);
    isqrt_kernel<<<(2 * N + T - 1) / T, T, 0, stream>>>(deg_out, 2 * N);

    // layer 1 (IN_DIM = 5)
    hipMemsetAsync(agg5, 0, (size_t)5 * N * sizeof(float), stream);
    spmm5_kernel<<<(E + T - 1) / T, T, 0, stream>>>(src, dst, features, deg_out, agg5, E);
    {
        long long total = (long long)N * HID;
        layer1_kernel<<<(unsigned)((total + T - 1) / T), T, 0, stream>>>(agg5, deg_in, W1, b1, h, N);
    }

    // layers 2..4: h -> agg (SpMM) -> h (WMMA GEMM + ReLU)
    const float* Ws[3] = { W2, W3, W4 };
    const float* bs[3] = { b2, b3, b4 };
    for (int l = 0; l < 3; ++l) {
        hipMemsetAsync(agg, 0, (size_t)HID * N * sizeof(float), stream);
        long long sp_threads = (long long)E * 32;
        spmm128_kernel<<<(unsigned)((sp_threads + T - 1) / T), T, 0, stream>>>(src, dst, h, deg_out, agg, E);
        gemm_relu_wmma<<<(N + 63) / 64, T, 0, stream>>>(agg, deg_in, Ws[l], bs[l], h, N);
    }

    // pooling + final projection
    hipMemsetAsync(gsums, 0, (size_t)(NG * HID + NG) * sizeof(float), stream);
    pool_kernel<<<(N + 127) / 128, 128, 0, stream>>>(h, gids, gsums, cnt, N);
    final_kernel<<<1, 256, 0, stream>>>(gsums, cnt, Wp, bp, out);
}
